// OptimizedVectorQuantizer_76544907149321
// MI455X (gfx1250) — compile-verified, matrix operand order checked by reading, not run on hardware
//
#include <hip/hip_runtime.h>
#include <hip/hip_bf16.h>
#include <stdint.h>

// ---------------------------------------------------------------------------
// VQ-VAE vector quantizer for MI455X (gfx1250, wave32, WMMA + TDM)
//   inputs  : [8, 2048, 256] fp32   (N = 16384 rows, D = 256)
//   codebook: [4096, 256]    fp32   (K = 4096)
//   out     : quantized fp32 [N*256] ++ indices i32 [N] ++ loss f32 [1]
//
// GEMM (x @ cb^T, 34.4 GFLOP) on v_wmma_f32_16x16x32_bf16, fp32 accumulate.
// Codebook chunks are DMA'd into LDS by the Tensor Data Mover (double
// buffered, TENSORcnt-tracked, TDM pad feature de-conflicts LDS banks).
// Each wave register-blocks M=32 rows (two A fragment sets) so every
// ds_load_b128 B fragment feeds two WMMAs -> LDS bandwidth ~= matrix rate.
// ---------------------------------------------------------------------------

#define VQ_D       256
#define VQ_K       4096
#define VQ_N       16384
#define VQ_KSTEPS  8        // 256 / 32 K-dim per wmma
#define CHUNK      32       // codes staged per TDM transfer (2 N-tiles)
#define NCHUNK     (VQ_K / CHUNK)
#define ROW_SH     264      // LDS row stride in shorts: 256 + 8 (TDM pad 16B)
#define WAVES      2        // waves per block
#define ROWS_W     32       // rows per wave (2 A-fragment sets)
#define BLOCK      (WAVES * 32)

typedef __attribute__((ext_vector_type(16))) __bf16 v16bf;
typedef __attribute__((ext_vector_type(8)))  float  v8f;
typedef __attribute__((ext_vector_type(4)))  unsigned int v4u;
typedef __attribute__((ext_vector_type(8)))  int v8i;
typedef __attribute__((ext_vector_type(4)))  int v4i;

#if defined(__has_builtin)
#if __has_builtin(__builtin_amdgcn_tensor_load_to_lds)
#define VQ_HAVE_TDM 1
#endif
#endif

union BFrag {
    v16bf v;
    unsigned int u[8];
};

static __device__ __forceinline__ unsigned int pack_bf16x2(float a, float b) {
    unsigned int ua = __float_as_uint(a);
    unsigned int ub = __float_as_uint(b);
    ua = (ua + 0x7FFFu + ((ua >> 16) & 1u)) >> 16;
    ub = (ub + 0x7FFFu + ((ub >> 16) & 1u)) >> 16;
    return (ua & 0xFFFFu) | (ub << 16);
}

static __device__ __forceinline__ unsigned short f2bf(float a) {
    unsigned int ua = __float_as_uint(a);
    ua = (ua + 0x7FFFu + ((ua >> 16) & 1u)) >> 16;
    return (unsigned short)ua;
}

static __device__ __forceinline__ void load_bfrag(BFrag& f,
                                                  const unsigned short* br,
                                                  int eoff) {
    const uint4* q0 = (const uint4*)(br + eoff);
    const uint4* q1 = (const uint4*)(br + eoff + 16);
    uint4 c0 = q0[0], c1 = q1[0];
    f.u[0] = c0.x; f.u[1] = c0.y; f.u[2] = c0.z; f.u[3] = c0.w;
    f.u[4] = c1.x; f.u[5] = c1.y; f.u[6] = c1.z; f.u[7] = c1.w;
}

#ifdef VQ_HAVE_TDM
// TDM transfer: CHUNK x 256 bf16 tile -> LDS with 16B pad per 512B row,
// so staged rows stride 528B (132 dwords == 4 mod 64 banks). D# per ISA §8.
static __device__ __forceinline__ void tdm_load_chunk(unsigned int lds_addr,
                                                      const unsigned short* gsrc)
{
    unsigned long long ga = (unsigned long long)(uintptr_t)gsrc;
    v4u g0;
    g0[0] = 1u;                                 // count=1 (valid user D#)
    g0[1] = lds_addr;                           // LDS dest byte address
    g0[2] = (unsigned int)(ga & 0xFFFFFFFFu);   // global_addr[31:0]
    g0[3] = (unsigned int)((ga >> 32) & 0x01FFFFFFu) | (2u << 30); // [56:32]|type=2

    v8i g1;
    g1[0] = (int)((1u << 16)        // data_size = 1 -> 2 bytes
                | (1u << 20)        // pad_enable
                | (6u << 22)        // pad_interval: 128 dwords (one 512B row)
                | (3u << 25));      // pad_amount:   4 dwords  (16B)
    g1[1] = (int)(256u << 16);      // tensor_dim0 = 256
    g1[2] = (int)((unsigned)CHUNK << 16); // tensor_dim1 = CHUNK
    g1[3] = (int)(256u << 16);      // tile_dim0 = 256
    g1[4] = (int)CHUNK;             // tile_dim1 = CHUNK; tile_dim2 = 0
    g1[5] = (int)256u;              // tensor_dim0_stride = 256
    g1[6] = 0;
    g1[7] = 0;

    v4i g2 = (v4i)0;
    v4i g3 = (v4i)0;
#if __has_include(<hip/amd_detail/amd_gfx1250_TDM.h>)
    v8i g4 = (v8i)0;
    __builtin_amdgcn_tensor_load_to_lds(g0, g1, g2, g3, g4, 0);
#else
    __builtin_amdgcn_tensor_load_to_lds(g0, g1, g2, g3, 0);
#endif
}
#endif

// ---------------------------------------------------------------------------
// Prep: codebook fp32 -> bf16 (WMMA B operand), ||e||^2 fp32, zero loss.
// ---------------------------------------------------------------------------
__global__ __launch_bounds__(256) void vq_prep_kernel(
    const float* __restrict__ cb, unsigned short* __restrict__ cb_bf,
    float* __restrict__ cbnorm, float* __restrict__ lossAcc)
{
    const int row = blockIdx.x;
    const int t   = threadIdx.x;
    const float f = cb[(size_t)row * VQ_D + t];
    cb_bf[(size_t)row * VQ_D + t] = f2bf(f);

    __shared__ float red[256];
    red[t] = f * f;
    __syncthreads();
    #pragma unroll
    for (int s = 128; s > 0; s >>= 1) {
        if (t < s) red[t] += red[t + s];
        __syncthreads();
    }
    if (t == 0) cbnorm[row] = red[0];
    if (row == 0 && t == 0) *lossAcc = 0.0f;
}

// ---------------------------------------------------------------------------
// Main: 2 waves/block, 32 rows per wave; codebook staged through LDS by TDM.
// ---------------------------------------------------------------------------
__global__ __launch_bounds__(BLOCK) void vq_wmma_kernel(
    const float* __restrict__ x, const float* __restrict__ cb,
    const unsigned short* __restrict__ cb_bf, const float* __restrict__ cbnorm,
    float* __restrict__ outq, int* __restrict__ outidx,
    float* __restrict__ lossAcc)
{
    const int lane  = threadIdx.x & 31;
    const int wave  = threadIdx.x >> 5;
    const int half  = lane >> 4;
    const int l15   = lane & 15;
    const int rowBase = (blockIdx.x * WAVES + wave) * ROWS_W;

    __shared__ __align__(16) unsigned short lbuf[2][CHUNK * ROW_SH];
    __shared__ int   s_idx[WAVES][ROWS_W];
    __shared__ float s_sum[WAVES];

    // ---- Resident bf16 A fragments: two 16x256 x-tiles (M blocking) ------
    BFrag afrag[2][VQ_KSTEPS];
    #pragma unroll
    for (int s = 0; s < 2; ++s) {
        const float* xrow = x + (size_t)(rowBase + s * 16 + l15) * VQ_D;
        #pragma unroll
        for (int kk = 0; kk < VQ_KSTEPS; ++kk) {
            const int k0 = kk * 32 + half * 8;
            const float4* p0 = (const float4*)(xrow + k0);
            const float4* p1 = (const float4*)(xrow + k0 + 16);
            float4 a0 = p0[0], a1 = p0[1];
            float4 b0 = p1[0], b1 = p1[1];
            afrag[s][kk].u[0] = pack_bf16x2(a0.x, a0.y);
            afrag[s][kk].u[1] = pack_bf16x2(a0.z, a0.w);
            afrag[s][kk].u[2] = pack_bf16x2(a1.x, a1.y);
            afrag[s][kk].u[3] = pack_bf16x2(a1.z, a1.w);
            afrag[s][kk].u[4] = pack_bf16x2(b0.x, b0.y);
            afrag[s][kk].u[5] = pack_bf16x2(b0.z, b0.w);
            afrag[s][kk].u[6] = pack_bf16x2(b1.x, b1.y);
            afrag[s][kk].u[7] = pack_bf16x2(b1.z, b1.w);
        }
    }

    // argmin slot (s,v) covers row m = s*16 + half*8 + v
    float minv[2][8];
    int   mini[2][8];
    #pragma unroll
    for (int s = 0; s < 2; ++s)
        #pragma unroll
        for (int v = 0; v < 8; ++v) { minv[s][v] = 3.4e38f; mini[s][v] = 0; }

    // ---- Stage chunk 0 ----------------------------------------------------
#ifdef VQ_HAVE_TDM
    if (wave == 0) {
        tdm_load_chunk((unsigned int)(uintptr_t)&lbuf[0][0], cb_bf);
        __builtin_amdgcn_s_wait_tensorcnt(0);
    }
#else
    for (int i = threadIdx.x; i < CHUNK * 32; i += BLOCK) {
        const int r = i >> 5, c = i & 31;
        const uint4* src = (const uint4*)(cb_bf + (size_t)r * VQ_D) + c;
        *(uint4*)((char*)&lbuf[0][0] + r * (ROW_SH * 2) + c * 16) = *src;
    }
#endif
    __syncthreads();

    // ---- Sweep codebook: compute chunk ch while TDM stages chunk ch+1 -----
    for (int ch = 0; ch < NCHUNK; ++ch) {
        const int cur = ch & 1;
        if (ch + 1 < NCHUNK) {
            const unsigned short* gsrc = cb_bf + (size_t)(ch + 1) * CHUNK * VQ_D;
#ifdef VQ_HAVE_TDM
            if (wave == 0)
                tdm_load_chunk((unsigned int)(uintptr_t)&lbuf[cur ^ 1][0], gsrc);
#else
            for (int i = threadIdx.x; i < CHUNK * 32; i += BLOCK) {
                const int r = i >> 5, c = i & 31;
                const uint4* src = (const uint4*)(gsrc + (size_t)r * VQ_D) + c;
                *(uint4*)((char*)&lbuf[cur ^ 1][0] + r * (ROW_SH * 2) + c * 16) = *src;
            }
#endif
        }

        const int nbBase = ch * CHUNK;
        const float cn0 = cbnorm[nbBase + l15];
        const float cn1 = cbnorm[nbBase + 16 + l15];
        v8f acc[2][2];
        #pragma unroll
        for (int s = 0; s < 2; ++s) {
            acc[s][0] = (v8f)(0.0f);
            acc[s][1] = (v8f)(0.0f);
        }
        const unsigned short* lb  = &lbuf[cur][0];
        const unsigned short* br0 = lb + (size_t)l15 * ROW_SH;        // tile 0
        const unsigned short* br1 = lb + (size_t)(16 + l15) * ROW_SH; // tile 1

        // software-pipelined B fragments: issue kk+1 loads before kk's WMMAs
        BFrag b0n, b1n;
        load_bfrag(b0n, br0, half * 8);
        load_bfrag(b1n, br1, half * 8);
        #pragma unroll
        for (int kk = 0; kk < VQ_KSTEPS; ++kk) {
            BFrag b0c = b0n, b1c = b1n;
            if (kk + 1 < VQ_KSTEPS) {
                const int eoff = (kk + 1) * 32 + half * 8;
                load_bfrag(b0n, br0, eoff);
                load_bfrag(b1n, br1, eoff);
            }
            acc[0][0] = __builtin_amdgcn_wmma_f32_16x16x32_bf16(
                false, afrag[0][kk].v, false, b0c.v, (short)0, acc[0][0], false, false);
            acc[0][1] = __builtin_amdgcn_wmma_f32_16x16x32_bf16(
                false, afrag[0][kk].v, false, b1c.v, (short)0, acc[0][1], false, false);
            acc[1][0] = __builtin_amdgcn_wmma_f32_16x16x32_bf16(
                false, afrag[1][kk].v, false, b0c.v, (short)0, acc[1][0], false, false);
            acc[1][1] = __builtin_amdgcn_wmma_f32_16x16x32_bf16(
                false, afrag[1][kk].v, false, b1c.v, (short)0, acc[1][1], false, false);
        }

        #pragma unroll
        for (int s = 0; s < 2; ++s) {
            #pragma unroll
            for (int v = 0; v < 8; ++v) {
                const float d0 = cn0 - 2.0f * acc[s][0][v];
                const float d1 = cn1 - 2.0f * acc[s][1][v];
                const int i0 = nbBase + l15;
                const int i1 = nbBase + 16 + l15;
                if (d0 < minv[s][v]) { minv[s][v] = d0; mini[s][v] = i0; }
                if (d1 < minv[s][v]) { minv[s][v] = d1; mini[s][v] = i1; }
            }
        }

#ifdef VQ_HAVE_TDM
        if (wave == 0) __builtin_amdgcn_s_wait_tensorcnt(0);
#endif
        __syncthreads();   // next buffer staged; this buffer free for rewrite
    }

    // ---- Cross-lane argmin within each 16-lane half (first-index ties) ----
    #pragma unroll
    for (int s = 0; s < 2; ++s) {
        #pragma unroll
        for (int v = 0; v < 8; ++v) {
            #pragma unroll
            for (int m = 1; m < 16; m <<= 1) {
                const float ov = __shfl_xor(minv[s][v], m, 32);
                const int   oi = __shfl_xor(mini[s][v], m, 32);
                if (ov < minv[s][v] || (ov == minv[s][v] && oi < mini[s][v])) {
                    minv[s][v] = ov; mini[s][v] = oi;
                }
            }
        }
    }

    if (l15 == 0) {
        #pragma unroll
        for (int s = 0; s < 2; ++s)
            #pragma unroll
            for (int v = 0; v < 8; ++v)
                s_idx[wave][s * 16 + half * 8 + v] = mini[s][v];
    }
    __syncthreads();

    // ---- Gather fp32 codebook rows, write output, accumulate loss --------
    // quantized_st = inputs + sg(quantized - inputs) == quantized numerically
    float sumsq = 0.0f;
    for (int r = 0; r < ROWS_W; ++r) {
        const int idx = s_idx[wave][r];
        const float4* crow = (const float4*)(cb + (size_t)idx * VQ_D);
        const float4* xr4  = (const float4*)(x + (size_t)(rowBase + r) * VQ_D);
        float4*       or4  = (float4*)(outq + (size_t)(rowBase + r) * VQ_D);
        const int c4 = lane * 2;
        #pragma unroll
        for (int j = 0; j < 2; ++j) {
            const float4 q  = crow[c4 + j];
            const float4 xv = xr4[c4 + j];
            or4[c4 + j] = q;
            const float dx = q.x - xv.x, dy = q.y - xv.y;
            const float dz = q.z - xv.z, dw = q.w - xv.w;
            sumsq += dx * dx + dy * dy + dz * dz + dw * dw;
        }
    }
    outidx[rowBase + lane] = s_idx[wave][lane];   // 32 indices, 32 lanes

    #pragma unroll
    for (int m = 16; m >= 1; m >>= 1) sumsq += __shfl_xor(sumsq, m, 32);
    if (lane == 0) s_sum[wave] = sumsq;
    __syncthreads();
    if (threadIdx.x == 0) {
        atomicAdd(lossAcc, s_sum[0] + s_sum[1]);
    }
}

__global__ void vq_finalize_kernel(const float* __restrict__ lossAcc,
                                   float* __restrict__ outLoss)
{
    *outLoss = 0.25f * (*lossAcc) / (float)((size_t)VQ_N * VQ_D);
}

// ---------------------------------------------------------------------------
extern "C" void kernel_launch(void* const* d_in, const int* in_sizes, int n_in,
                              void* d_out, int out_size, void* d_ws, size_t ws_size,
                              hipStream_t stream) {
    (void)in_sizes; (void)n_in; (void)out_size; (void)ws_size;

    const float* x  = (const float*)d_in[0];   // [N, 256] fp32
    const float* cb = (const float*)d_in[1];   // [K, 256] fp32

    // workspace: bf16 codebook (2 MB) | cbnorm (16 KB) | loss accumulator
    unsigned short* cb_bf  = (unsigned short*)d_ws;
    float*          cbnorm = (float*)((char*)d_ws + (size_t)VQ_K * VQ_D * sizeof(unsigned short));
    float*          lossAcc = cbnorm + VQ_K;

    float* outq    = (float*)d_out;
    int*   outidx  = (int*)(outq + (size_t)VQ_N * VQ_D);
    float* outLoss = (float*)(outq + (size_t)VQ_N * VQ_D + VQ_N);

    vq_prep_kernel<<<VQ_K, 256, 0, stream>>>(cb, cb_bf, cbnorm, lossAcc);
    vq_wmma_kernel<<<VQ_N / (WAVES * ROWS_W), BLOCK, 0, stream>>>(
        x, cb, cb_bf, cbnorm, outq, outidx, lossAcc);
    vq_finalize_kernel<<<1, 1, 0, stream>>>(lossAcc, outLoss);
}
